// FNO_multi_2594160247371
// MI455X (gfx1250) — matrix operand order repaired and checked by reading, not compiled, not run on hardware
//
#include <hip/hip_runtime.h>
#include <math.h>

typedef __attribute__((ext_vector_type(16))) __bf16 v16bf;
typedef __attribute__((ext_vector_type(8)))  float  v8f;
typedef __attribute__((ext_vector_type(4)))  unsigned v4u;
typedef __attribute__((ext_vector_type(8)))  int      v8i;
typedef __attribute__((ext_vector_type(4)))  int      v4i;

#define B_      4
#define V_      3
#define SX_     256
#define SY_     256
#define TIN_    10
#define W_      32
#define M2_     16
#define VOUT_   3
#define STEP_   10
#define NPIX_   65536
#define NPLANE_ 384          // B_*W_*V_

// Native RNE float->bf16 conversion; lets the backend use v_cvt hardware paths.
__device__ __forceinline__ __bf16 f2bf(float f) { return (__bf16)f; }
__device__ __forceinline__ float gelu(float x) { return 0.5f * x * (1.0f + erff(x * 0.70710678118f)); }

// 16-bit A-matrix 16x32 layout (contiguous k, matches B fragments): lane,slot -> (m,k = (lane>>4)*16+slot)
__device__ __forceinline__ void amap(int lane, int s, int& m, int& k) {
  int hi = lane >> 4;
  m = lane & 15;
  k = hi * 16 + s;
}

__device__ __forceinline__ v8f wmma_bf(v16bf a, v16bf b, v8f c) {
  return __builtin_amdgcn_wmma_f32_16x16x32_bf16(false, a, false, b, (short)0, c, false, false);
}
__device__ __forceinline__ v16bf ldfrag(const __bf16* p) {
  return *reinterpret_cast<const v16bf*>(p);
}

// ---- Tensor Data Mover: 2D tile (tile0 contiguous elems x tile1 rows) -> LDS.
// D# per CDNA5 ISA 8.3-8.6; amdgpu-toolchain 6-arg builtin (v4u,v8i,v4i,v4i,v8i,i32).
__device__ __forceinline__ void tdm_load_tile(unsigned lds_off, unsigned long long ga,
                                              unsigned tile0, unsigned tile1,
                                              unsigned long long stride0) {
  v4u g0;
  g0[0] = 1u;                                              // count=1, user descriptor
  g0[1] = lds_off;                                         // LDS byte address
  g0[2] = (unsigned)ga;                                    // global_addr[31:0]
  g0[3] = ((unsigned)(ga >> 32) & 0x01FFFFFFu) | (2u << 30);  // ga[56:32] | type=2
  unsigned td0 = 1u << 16;                                 // tensor_dim0 (OOB bound)
  unsigned td1 = tile1;                                    // tensor_dim1
  v8i g1;
  g1[0] = (int)(2u << 16);                                 // data_size=4B
  g1[1] = (int)((td0 & 0xFFFFu) << 16);
  g1[2] = (int)((td0 >> 16) | ((td1 & 0xFFFFu) << 16));
  g1[3] = (int)((td1 >> 16) | (tile0 << 16));              // | tile_dim0
  g1[4] = (int)tile1;                                      // tile_dim1, tile_dim2=0
  g1[5] = (int)(stride0 & 0xFFFFFFFFu);
  g1[6] = (int)((stride0 >> 32) & 0xFFFFu);
  g1[7] = 0;
  v4i g2; g2[0] = g2[1] = g2[2] = g2[3] = 0;
  v4i g3; g3[0] = g3[1] = g3[2] = g3[3] = 0;
  v8i g4;
  g4[0] = g4[1] = g4[2] = g4[3] = g4[4] = g4[5] = g4[6] = g4[7] = 0;
  __builtin_amdgcn_tensor_load_to_lds(g0, g1, g2, g3, g4, 0);
}

// ------------- twiddle tables, pre-swizzled into WMMA fragment order -------------
// B-fragment: frag[tile][lane][slot] = B[k = (lane>>4)*16+slot][n = lane&15]
// A-fragment: frag[tile][lane][slot] = A[m][k] with (m,k) = amap(lane,slot)
__global__ void k_init_tw(__bf16* FyR, __bf16* FyI,     // 8*512  (y-DFT B)
                          __bf16* Fx,                   // 2*16*512 (x-DFT A: R then I)
                          __bf16* Gx,                   // 2*16*512 (x-iDFT A: R then I)
                          __bf16* Cy) {                 // 16*512 (y-iDFT B)
  const float TP = 6.283185307179586f;
  for (int i = threadIdx.x; i < 8 * 512; i += blockDim.x) {
    int kc = i >> 9, r = i & 511, lane = r >> 4, s = r & 15;
    int y = kc * 32 + ((lane >> 4) << 4) + s;
    int ky = lane & 15;
    float th = TP * (float)((y * ky) % SY_) / (float)SY_;
    FyR[i] = f2bf(cosf(th));
    FyI[i] = f2bf(-sinf(th));
  }
  for (int i = threadIdx.x; i < 16 * 512; i += blockDim.x) {
    int t = i >> 9, r = i & 511, lane = r >> 4, s = r & 15;
    int mt = t >> 3, kc = t & 7;
    int m, k; amap(lane, s, m, k);
    int j = mt * 16 + m;
    int f = (j < 16) ? j : (j + 224);
    int x = kc * 32 + k;
    float th = TP * (float)((f * x) % SX_) / (float)SX_;
    Fx[i]            = f2bf(cosf(th));     // real part
    Fx[i + 16 * 512] = f2bf(-sinf(th));    // imag part
  }
  for (int i = threadIdx.x; i < 16 * 512; i += blockDim.x) {
    int xt = i >> 9, r = i & 511, lane = r >> 4, s = r & 15;
    int m, k; amap(lane, s, m, k);
    int x = xt * 16 + m;
    int f = (k < 16) ? k : (k + 224);
    float th = TP * (float)((f * x) % SX_) / (float)SX_;
    Gx[i]            = f2bf(cosf(th));
    Gx[i + 16 * 512] = f2bf(sinf(th));     // inverse: +i
  }
  for (int i = threadIdx.x; i < 16 * 512; i += blockDim.x) {
    int yt = i >> 9, r = i & 511, lane = r >> 4, s = r & 15;
    int kk = ((lane >> 4) << 4) + s;
    int y = yt * 16 + (lane & 15);
    int ky = (kk < 16) ? kk : (kk - 16);
    float sc = ((ky == 0) ? 1.0f : 2.0f) / 65536.0f;
    float th = TP * (float)((y * ky) % SY_) / (float)SY_;
    Cy[i] = f2bf((kk < 16) ? (sc * cosf(th)) : (-sc * sinf(th)));
  }
}

// Conv weights (W[o][c], 6 layers) -> bf16 B-fragment tables [layer][ntile][lane][slot]
__global__ void k_init_wf(const float* __restrict__ W, __bf16* __restrict__ F) {
  int i = blockIdx.x * blockDim.x + threadIdx.x;   // 6*2*512
  if (i >= 6 * 2 * 512) return;
  int t = i >> 9, r = i & 511, lane = r >> 4, s = r & 15;
  int l = t >> 1, nt = t & 1;
  int n = lane & 15;
  int k = ((lane >> 4) << 4) + s;
  F[i] = f2bf(W[(size_t)l * 1024 + (size_t)(nt * 16 + n) * W_ + k]);
}

// ---------------------------------------------------------------- lifting
__global__ void k_fc0(const float* __restrict__ x, const float* __restrict__ w,
                      const float* __restrict__ b, float* __restrict__ h) {
  int idx = blockIdx.x * blockDim.x + threadIdx.x;
  if (idx >= B_ * V_ * NPIX_) return;
  int p = idx % NPIX_;
  int bv = idx / NPIX_;
  int v = bv % V_, bb = bv / V_;
  float in[12];
  const float* xp = x + (size_t)idx * TIN_;
#pragma unroll
  for (int t = 0; t < TIN_; ++t) in[t] = xp[t];
  in[10] = (float)(p >> 8) * (1.0f / 255.0f);
  in[11] = (float)(p & 255) * (1.0f / 255.0f);
  for (int c = 0; c < W_; ++c) {
    float s = b[c];
#pragma unroll
    for (int t = 0; t < 12; ++t) s += in[t] * w[t * W_ + c];
    h[(size_t)((bb * W_ + c) * V_ + v) * NPIX_ + p] = s;
  }
}

// ---------------------------------------------------------------- y-DFT (WMMA)
__global__ void k_dfty(const float* __restrict__ h,
                       const __bf16* __restrict__ FyR, const __bf16* __restrict__ FyI,
                       float* __restrict__ TyR, float* __restrict__ TyI) {
  int wave = (blockIdx.x * blockDim.x + threadIdx.x) >> 5;  // NPLANE*16 waves
  int lane = threadIdx.x & 31;
  int plane = wave >> 4;
  int x0 = (wave & 15) << 4;
  const float* src = h + (size_t)plane * NPIX_;
  v8f accR = {}, accI = {};
  for (int kc = 0; kc < 8; ++kc) {
    if (kc < 7)
      __builtin_prefetch(src + (x0 + (lane & 15)) * SY_ + (kc + 1) * 32, 0, 1);
    v16bf a;
#pragma unroll
    for (int hp = 0; hp < 8; ++hp) {                // slot pairs share consecutive k
      int m, k0; amap(lane, hp * 2, m, k0);
      float2 pv = *reinterpret_cast<const float2*>(&src[(x0 + m) * SY_ + kc * 32 + k0]);
      a[hp * 2]     = f2bf(pv.x);
      a[hp * 2 + 1] = f2bf(pv.y);
    }
    v16bf br = ldfrag(FyR + (kc * 32 + lane) * 16);
    v16bf bi = ldfrag(FyI + (kc * 32 + lane) * 16);
    accR = wmma_bf(a, br, accR);
    accI = wmma_bf(a, bi, accI);
  }
  int n = lane & 15, hi = lane >> 4;
#pragma unroll
  for (int r = 0; r < 8; ++r) {
    int m = hi * 8 + r;
    size_t o = ((size_t)plane * SX_ + (x0 + m)) * M2_ + n;
    TyR[o] = accR[r];
    TyI[o] = accI[r];
  }
}

// ---------------------------------------------------------------- x-DFT (complex, WMMA)
__global__ void k_dftx(const float* __restrict__ TyR, const float* __restrict__ TyI,
                       const __bf16* __restrict__ Fx,
                       float* __restrict__ ZR, float* __restrict__ ZI) {
  int wave = (blockIdx.x * blockDim.x + threadIdx.x) >> 5;  // NPLANE*2 waves
  int lane = threadIdx.x & 31;
  int plane = wave >> 1;
  int mt = wave & 1;
  v8f arr = {}, aii = {}, ari = {}, air = {};
  for (int kc = 0; kc < 8; ++kc) {
    v16bf Ar = ldfrag(Fx + ((mt * 8 + kc) * 32 + lane) * 16);
    v16bf Ai = ldfrag(Fx + (16 * 512) + ((mt * 8 + kc) * 32 + lane) * 16);
    v16bf Br, Bi;
#pragma unroll
    for (int s = 0; s < 16; ++s) {
      int kb = ((lane >> 4) << 4) + s;
      int n = lane & 15;
      size_t bo = ((size_t)plane * SX_ + kc * 32 + kb) * M2_ + n;
      Br[s] = f2bf(TyR[bo]);
      Bi[s] = f2bf(TyI[bo]);
    }
    arr = wmma_bf(Ar, Br, arr);
    aii = wmma_bf(Ai, Bi, aii);
    ari = wmma_bf(Ar, Bi, ari);
    air = wmma_bf(Ai, Br, air);
  }
  int n = lane & 15, hi = lane >> 4;
#pragma unroll
  for (int r = 0; r < 8; ++r) {
    int m = hi * 8 + r;
    size_t o = ((size_t)plane * 32 + mt * 16 + m) * M2_ + n;
    ZR[o] = arr[r] - aii[r];
    ZI[o] = ari[r] + air[r];
  }
}

// ---------------------------------------------------------------- spectral channel mix (VALU)
__global__ void k_mix(const float* __restrict__ ZR, const float* __restrict__ ZI,
                      const float* __restrict__ w1r, const float* __restrict__ w1i,
                      const float* __restrict__ w2r, const float* __restrict__ w2i,
                      int layer, float* __restrict__ ZmR, float* __restrict__ ZmI) {
  int idx = blockIdx.x * blockDim.x + threadIdx.x;  // B*W*V*32*16
  if (idx >= B_ * W_ * V_ * 32 * M2_) return;
  int ky = idx % 16; int t = idx / 16;
  int kx = t % 32; t /= 32;
  int v = t % V_; t /= V_;
  int o = t % W_; int b = t / W_;
  const float *wr, *wi; int kxm;
  if (kx < 16) { wr = w1r; wi = w1i; kxm = kx; }
  else         { wr = w2r; wi = w2i; kxm = kx - 16; }
  size_t wb = (size_t)layer * 786432 + (size_t)o * 768 + (size_t)v * 256 + kxm * 16 + ky;
  size_t zb = (size_t)b * 49152 + (size_t)v * 512 + kx * 16 + ky;
  float sr = 0.f, si = 0.f;
  for (int i = 0; i < 32; ++i) {
    float xr = ZR[zb + (size_t)i * 1536], xi = ZI[zb + (size_t)i * 1536];
    float ar = wr[wb + (size_t)i * 24576], ai = wi[wb + (size_t)i * 24576];
    sr += xr * ar - xi * ai;
    si += xr * ai + xi * ar;
  }
  size_t ob = (((size_t)(b * W_ + o) * V_ + v) * 32 + kx) * M2_ + ky;
  ZmR[ob] = sr;
  ZmI[ob] = si;
}

// ---------------------------------------------------------------- inverse x-DFT (complex, WMMA)
__global__ void k_idftx(const float* __restrict__ ZmR, const float* __restrict__ ZmI,
                        const __bf16* __restrict__ Gx,
                        float* __restrict__ SxR, float* __restrict__ SxI) {
  int wave = (blockIdx.x * blockDim.x + threadIdx.x) >> 5;  // NPLANE*16 waves
  int lane = threadIdx.x & 31;
  int plane = wave >> 4;
  int xt = wave & 15;
  v16bf Ar = ldfrag(Gx + (xt * 32 + lane) * 16);
  v16bf Ai = ldfrag(Gx + (16 * 512) + (xt * 32 + lane) * 16);
  v16bf Br, Bi;
#pragma unroll
  for (int s = 0; s < 16; ++s) {
    int kb = ((lane >> 4) << 4) + s;
    int n = lane & 15;
    size_t bo = ((size_t)plane * 32 + kb) * M2_ + n;
    Br[s] = f2bf(ZmR[bo]);
    Bi[s] = f2bf(ZmI[bo]);
  }
  v8f arr = {}, aii = {}, ari = {}, air = {};
  arr = wmma_bf(Ar, Br, arr);
  aii = wmma_bf(Ai, Bi, aii);
  ari = wmma_bf(Ar, Bi, ari);
  air = wmma_bf(Ai, Br, air);
  int n = lane & 15, hi = lane >> 4;
#pragma unroll
  for (int r = 0; r < 8; ++r) {
    int m = hi * 8 + r;
    size_t o = ((size_t)plane * SX_ + xt * 16 + m) * M2_ + n;
    SxR[o] = arr[r] - aii[r];
    SxI[o] = ari[r] + air[r];
  }
}

// ---------------------------------------------------------------- inverse y-DFT to real (WMMA)
__global__ void k_idfty(const float* __restrict__ SxR, const float* __restrict__ SxI,
                        const __bf16* __restrict__ Cy, float* __restrict__ X1) {
  int wave = (blockIdx.x * blockDim.x + threadIdx.x) >> 5;  // NPLANE*16 waves
  int lane = threadIdx.x & 31;
  int plane = wave >> 4;
  int xt = wave & 15;
  v16bf A;
#pragma unroll
  for (int hp = 0; hp < 8; ++hp) {                 // slot pairs: contiguous k
    int m, k0; amap(lane, hp * 2, m, k0);
    size_t ro = ((size_t)plane * SX_ + xt * 16 + m) * M2_;
    const float* bp = (k0 < 16) ? (SxR + ro + k0) : (SxI + ro + k0 - 16);
    float2 pv = *reinterpret_cast<const float2*>(bp);
    A[hp * 2]     = f2bf(pv.x);
    A[hp * 2 + 1] = f2bf(pv.y);
  }
  int n = lane & 15, hi = lane >> 4;
  for (int yt = 0; yt < 16; ++yt) {
    v16bf Bv = ldfrag(Cy + (yt * 32 + lane) * 16);
    v8f acc = {};
    acc = wmma_bf(A, Bv, acc);
#pragma unroll
    for (int r = 0; r < 8; ++r) {
      int m = hi * 8 + r;
      X1[(size_t)plane * NPIX_ + (size_t)(xt * 16 + m) * SY_ + yt * 16 + n] = acc[r];
    }
  }
}

// ------------------- conv m1 + GELU (TDM-staged A tile in LDS, WMMA, in place)
__global__ void k_cgelu(float* __restrict__ X1, const __bf16* __restrict__ Fm1,
                        const float* __restrict__ m1b, int layer) {
  __shared__ float smem[W_ * 64];                // 32 channel rows x 64 pixels
  int wg = blockIdx.x;                           // B*V*1024 blocks, 128 thr
  int bv = wg >> 10;
  int p0 = (wg & 1023) << 6;
  int b = bv / V_, v = bv % V_;
  const float* gsrc = X1 + ((size_t)(b * W_) * V_ + v) * NPIX_ + p0;  // channel-0 row
  if (threadIdx.x < 32) {
    tdm_load_tile((unsigned)(unsigned long long)(void*)smem,
                  (unsigned long long)(const void*)gsrc,
                  64u, (unsigned)W_, (unsigned long long)(V_ * NPIX_));
    __builtin_amdgcn_s_wait_tensorcnt(0);
  }
  __syncthreads();
  int wv = threadIdx.x >> 5;                     // sub-tile: 16 pixels
  int lane = threadIdx.x & 31;
  v16bf A;
#pragma unroll
  for (int s = 0; s < 16; ++s) {
    int m, k; amap(lane, s, m, k);
    A[s] = f2bf(smem[k * 64 + wv * 16 + m]);
  }
  const float* Bb = m1b + layer * W_;
  int n = lane & 15, hi = lane >> 4;
  for (int nt = 0; nt < 2; ++nt) {
    v16bf Bv = ldfrag(Fm1 + ((layer * 2 + nt) * 32 + lane) * 16);
    v8f acc = {};
    acc = wmma_bf(A, Bv, acc);
    int o = nt * 16 + n;
    float bias = Bb[o];
#pragma unroll
    for (int r = 0; r < 8; ++r) {
      int m = hi * 8 + r;
      X1[((size_t)(b * W_ + o) * V_ + v) * NPIX_ + p0 + wv * 16 + m] = gelu(acc[r] + bias);
    }
  }
}

// ---- conv m2 + conv w + grid bias + GELU + residual (two TDM-staged tiles, WMMA)
__global__ void k_ccomb(const float* __restrict__ T1, const float* __restrict__ H,
                        const float* __restrict__ res,
                        const __bf16* __restrict__ Fm2, const float* __restrict__ m2b,
                        const __bf16* __restrict__ Fw, const float* __restrict__ wb,
                        const float* __restrict__ bw, const float* __restrict__ bb,
                        int layer, float* __restrict__ out) {
  __shared__ float sT[W_ * 64];
  __shared__ float sH[W_ * 64];
  int wg = blockIdx.x;
  int bv = wg >> 10;
  int p0 = (wg & 1023) << 6;
  int b = bv / V_, v = bv % V_;
  size_t base = ((size_t)(b * W_) * V_ + v) * NPIX_ + p0;
  if (threadIdx.x < 32) {
    tdm_load_tile((unsigned)(unsigned long long)(void*)sT,
                  (unsigned long long)(const void*)(T1 + base),
                  64u, (unsigned)W_, (unsigned long long)(V_ * NPIX_));
    tdm_load_tile((unsigned)(unsigned long long)(void*)sH,
                  (unsigned long long)(const void*)(H + base),
                  64u, (unsigned)W_, (unsigned long long)(V_ * NPIX_));
    __builtin_amdgcn_s_wait_tensorcnt(0);
  }
  __syncthreads();
  int wv = threadIdx.x >> 5;
  int lane = threadIdx.x & 31;
  v16bf A1, A2;
#pragma unroll
  for (int s = 0; s < 16; ++s) {
    int m, k; amap(lane, s, m, k);
    A1[s] = f2bf(sT[k * 64 + wv * 16 + m]);
    A2[s] = f2bf(sH[k * 64 + wv * 16 + m]);
  }
  int n = lane & 15, hi = lane >> 4;
  for (int nt = 0; nt < 2; ++nt) {
    v16bf Bm  = ldfrag(Fm2 + ((layer * 2 + nt) * 32 + lane) * 16);
    v16bf Bw2 = ldfrag(Fw  + ((layer * 2 + nt) * 32 + lane) * 16);
    v8f acc = {};
    acc = wmma_bf(A1, Bm, acc);
    acc = wmma_bf(A2, Bw2, acc);
    int o = nt * 16 + n;
    float c0  = m2b[layer * W_ + o] + wb[layer * W_ + o] + bb[layer * W_ + o];
    float bwx = bw[(layer * W_ + o) * 2 + 0];
    float bwy = bw[(layer * W_ + o) * 2 + 1];
#pragma unroll
    for (int r = 0; r < 8; ++r) {
      int m = hi * 8 + r;
      int p = p0 + wv * 16 + m;
      float gx = (float)(p >> 8) * (1.0f / 255.0f);
      float gy = (float)(p & 255) * (1.0f / 255.0f);
      float val = gelu(acc[r] + c0 + bwx * gx + bwy * gy);
      size_t oi = ((size_t)(b * W_ + o) * V_ + v) * NPIX_ + p;
      if (res) val += res[oi];
      out[oi] = val;
    }
  }
}

// ---------------------------------------------------------------- fc1v + GELU
__global__ void k_fc1v(const float* __restrict__ H, const float* __restrict__ w,
                       const float* __restrict__ bias, float* __restrict__ hv) {
  int idx = blockIdx.x * blockDim.x + threadIdx.x;
  if (idx >= B_ * NPIX_ * W_) return;
  int c = idx % W_;
  int t = idx / W_;
  int p = t % NPIX_;
  int b = t / NPIX_;
  float hval[V_];
#pragma unroll
  for (int v = 0; v < V_; ++v)
    hval[v] = H[((size_t)(b * W_ + c) * V_ + v) * NPIX_ + p];
#pragma unroll
  for (int vo = 0; vo < VOUT_; ++vo) {
    float s = bias[vo];
#pragma unroll
    for (int v = 0; v < V_; ++v) s += hval[v] * w[v * VOUT_ + vo];
    hv[((size_t)(b * VOUT_ + vo) * NPIX_ + p) * W_ + c] = gelu(s);
  }
}

// ---------------------------------------------------------------- head: fc1t+GELU+fc2t
__global__ void k_head(const float* __restrict__ hv, const float* __restrict__ w1,
                       const float* __restrict__ b1, const float* __restrict__ w2,
                       const float* __restrict__ b2, float* __restrict__ out) {
  __shared__ float sW1[W_ * 256];
  __shared__ float sW2[256 * STEP_];
  __shared__ float sB1[256];
  for (int i = threadIdx.x; i < W_ * 256; i += blockDim.x) sW1[i] = w1[i];
  for (int i = threadIdx.x; i < 256 * STEP_; i += blockDim.x) sW2[i] = w2[i];
  for (int i = threadIdx.x; i < 256; i += blockDim.x) sB1[i] = b1[i];
  __syncthreads();
  int idx = blockIdx.x * blockDim.x + threadIdx.x;
  if (idx >= B_ * VOUT_ * NPIX_) return;
  const float* vin = hv + (size_t)idx * W_;
  float vec[W_];
#pragma unroll
  for (int c = 0; c < W_; ++c) vec[c] = vin[c];
  float o[STEP_];
#pragma unroll
  for (int k = 0; k < STEP_; ++k) o[k] = b2[k];
  for (int j = 0; j < 256; ++j) {
    float s = sB1[j];
#pragma unroll
    for (int c = 0; c < W_; ++c) s += vec[c] * sW1[c * 256 + j];
    float t = gelu(s);
#pragma unroll
    for (int k = 0; k < STEP_; ++k) o[k] += t * sW2[j * STEP_ + k];
  }
  float* op = out + (size_t)idx * STEP_;
#pragma unroll
  for (int k = 0; k < STEP_; ++k) op[k] = o[k];
}

// ================================================================ launch
extern "C" void kernel_launch(void* const* d_in, const int* in_sizes, int n_in,
                              void* d_out, int out_size, void* d_ws, size_t ws_size,
                              hipStream_t stream) {
  (void)in_sizes; (void)n_in; (void)out_size; (void)ws_size;
  const float* x      = (const float*)d_in[0];
  const float* fc0_w  = (const float*)d_in[1];
  const float* fc0_b  = (const float*)d_in[2];
  const float* w1r    = (const float*)d_in[3];
  const float* w1i    = (const float*)d_in[4];
  const float* w2r    = (const float*)d_in[5];
  const float* w2i    = (const float*)d_in[6];
  const float* m1w    = (const float*)d_in[7];
  const float* m1b    = (const float*)d_in[8];
  const float* m2w    = (const float*)d_in[9];
  const float* m2b    = (const float*)d_in[10];
  const float* ww     = (const float*)d_in[11];
  const float* wb     = (const float*)d_in[12];
  const float* bw     = (const float*)d_in[13];
  const float* bb     = (const float*)d_in[14];
  const float* fc1v_w = (const float*)d_in[15];
  const float* fc1v_b = (const float*)d_in[16];
  const float* fc1t_w = (const float*)d_in[17];
  const float* fc1t_b = (const float*)d_in[18];
  const float* fc2t_w = (const float*)d_in[19];
  const float* fc2t_b = (const float*)d_in[20];

  char* ws = (char*)d_ws;
  size_t off = 0;
  auto carve = [&](size_t bytes) -> char* {
    char* p = ws + off;
    off += (bytes + 255) & ~(size_t)255;
    return p;
  };
  const size_t NH = (size_t)NPLANE_ * NPIX_;
  float* bufA = (float*)carve(NH * 4);
  float* bufR = (float*)carve(NH * 4);
  float* x1   = (float*)carve(NH * 4);
  const size_t NT = (size_t)NPLANE_ * SX_ * M2_;
  float* TyR = (float*)carve(NT * 4);
  float* TyI = (float*)carve(NT * 4);
  float* SxR = (float*)carve(NT * 4);
  float* SxI = (float*)carve(NT * 4);
  const size_t NZ = (size_t)NPLANE_ * 32 * M2_;
  float* ZR  = (float*)carve(NZ * 4);
  float* ZI  = (float*)carve(NZ * 4);
  float* ZmR = (float*)carve(NZ * 4);
  float* ZmI = (float*)carve(NZ * 4);
  __bf16* FyR = (__bf16*)carve(8 * 512 * 2);
  __bf16* FyI = (__bf16*)carve(8 * 512 * 2);
  __bf16* Fx  = (__bf16*)carve(2 * 16 * 512 * 2);   // R then I halves
  __bf16* Gx  = (__bf16*)carve(2 * 16 * 512 * 2);   // R then I halves
  __bf16* Cy  = (__bf16*)carve(16 * 512 * 2);
  __bf16* Fm1 = (__bf16*)carve(6 * 2 * 512 * 2);
  __bf16* Fm2 = (__bf16*)carve(6 * 2 * 512 * 2);
  __bf16* Fw  = (__bf16*)carve(6 * 2 * 512 * 2);

  k_init_tw<<<1, 256, 0, stream>>>(FyR, FyI, Fx, Gx, Cy);
  k_init_wf<<<24, 256, 0, stream>>>(m1w, Fm1);
  k_init_wf<<<24, 256, 0, stream>>>(m2w, Fm2);
  k_init_wf<<<24, 256, 0, stream>>>(ww, Fw);
  k_fc0<<<(B_ * V_ * NPIX_ + 255) / 256, 256, 0, stream>>>(x, fc0_w, fc0_b, bufA);

  auto layer = [&](int l, const float* src, float* dst, const float* res) {
    k_dfty<<<1536, 128, 0, stream>>>(src, FyR, FyI, TyR, TyI);
    k_dftx<<<192, 128, 0, stream>>>(TyR, TyI, Fx, ZR, ZI);
    k_mix<<<768, 256, 0, stream>>>(ZR, ZI, w1r, w1i, w2r, w2i, l, ZmR, ZmI);
    k_idftx<<<1536, 128, 0, stream>>>(ZmR, ZmI, Gx, SxR, SxI);
    k_idfty<<<1536, 128, 0, stream>>>(SxR, SxI, Cy, x1);
    k_cgelu<<<12288, 128, 0, stream>>>(x1, Fm1, m1b, l);
    k_ccomb<<<12288, 128, 0, stream>>>(x1, src, res, Fm2, m2b, Fw, wb, bw, bb, l, dst);
  };

  layer(0, bufA, bufR, nullptr);   // h0 = layer(0, h)
  layer(1, bufR, bufA, nullptr);   // h  = layer(1, h0)
  layer(2, bufA, bufA, bufR);      // h  = layer(2, h) + h0
  layer(3, bufA, bufR, nullptr);   // h1 = layer(3, h)
  layer(4, bufR, bufA, nullptr);   // h  = layer(4, h1)
  layer(5, bufA, bufA, bufR);      // h  = layer(5, h) + h1

  float* hv = x1;  // reuse spectral scratch
  k_fc1v<<<(B_ * NPIX_ * W_ + 255) / 256, 256, 0, stream>>>(bufA, fc1v_w, fc1v_b, hv);
  k_head<<<(B_ * VOUT_ * NPIX_ + 255) / 256, 256, 0, stream>>>(hv, fc1t_w, fc1t_b,
                                                               fc2t_w, fc2t_b, (float*)d_out);
}
